// MarginalLoss_3702261809485
// MI455X (gfx1250) — matrix operand order repaired and checked by reading, not compiled, not run on hardware
//
#include <hip/hip_runtime.h>
#include <hip/hip_bf16.h>

typedef __attribute__((ext_vector_type(16))) _Float16 v16h;
typedef __attribute__((ext_vector_type(8)))  _Float16 v8h;
typedef __attribute__((ext_vector_type(8)))  float    v8f;

namespace {
constexpr int   kB     = 8192;   // batch
constexpr int   kD     = 512;    // feature dim
constexpr float kTheta = 1.2f;
constexpr float kXi    = 0.3f;

constexpr int BTM  = 128;        // block tile rows
constexpr int BTN  = 256;        // block tile cols
constexpr int KT   = 32;         // k-step per WMMA pass
constexpr int NKT  = kD / KT;    // 16 k-steps
constexpr int LDSS = 40;         // LDS row stride in halves (32 + 8 pad, 16B-aligned rows)
}

// ---------------------------------------------------------------------------
// Kernel 0: zero the scalar output (harness poisons d_out).
// ---------------------------------------------------------------------------
__global__ void zero_out_kernel(float* out) {
    if (threadIdx.x == 0 && blockIdx.x == 0) out[0] = 0.0f;
}

// ---------------------------------------------------------------------------
// Kernel 1: per-row L2 normalize, f32 -> f16. One block (256 thr) per row.
// ---------------------------------------------------------------------------
__global__ __launch_bounds__(256)
void row_normalize_f16(const float* __restrict__ x, _Float16* __restrict__ nrm) {
    const int row = blockIdx.x;
    const int tid = threadIdx.x;
    const float* xr = x + (size_t)row * kD;

    float s = 0.0f;
    #pragma unroll
    for (int c = tid; c < kD; c += 256) { const float v = xr[c]; s += v * v; }

    __shared__ float red[256];
    red[tid] = s;
    __syncthreads();
    #pragma unroll
    for (int off = 128; off > 0; off >>= 1) {
        if (tid < off) red[tid] += red[tid + off];
        __syncthreads();
    }
    const float inv = rsqrtf(red[0]);

    _Float16* nr = nrm + (size_t)row * kD;
    #pragma unroll
    for (int c = tid; c < kD; c += 256) nr[c] = (_Float16)(xr[c] * inv);
}

// ---------------------------------------------------------------------------
// Kernel 2: fused Gram (n @ n^T) + hinge/indicator/mask + global reduction.
// 128x256 block tile, 8 wave32 waves in 2x4; each wave owns 64x64 = 4x4 WMMA
// accumulators. Global->LDS staging uses the CDNA5 async-to-LDS path
// (global_load_async_to_lds_b128, ASYNCcnt), double buffered. Per k-step all
// 16 fragment ds_load_b128s are issued before the 16-WMMA burst so the
// scheduler can use partial s_wait_dscnt interleaving.
// ---------------------------------------------------------------------------
__global__ __launch_bounds__(256)
void gram_hinge_wmma(const _Float16* __restrict__ nrm,
                     const int* __restrict__ y,
                     float* __restrict__ out) {
    const int bi = blockIdx.y;           // row block:  rows [bi*128, +128)
    const int bj = blockIdx.x;           // col block:  cols [bj*256, +256)
    if (bi > 2 * bj + 1) return;         // entire block strictly below diagonal

    __shared__ __attribute__((aligned(16))) _Float16 As[2][BTM][LDSS];
    __shared__ __attribute__((aligned(16))) _Float16 Bs[2][BTN][LDSS];
    __shared__ float red[256];
    __shared__ int   yl[BTM + BTN];

    const int tid  = threadIdx.x;
    const int lane = tid & 31;
    const int wid  = tid >> 5;           // 0..7
    const int wm   = wid & 1;            // 0..1 -> 64-row slab
    const int wn   = wid >> 1;           // 0..3 -> 64-col slab

    const int i0 = bi * BTM;
    const int j0 = bj * BTN;

    v8f acc[4][4];
    const v8f vzero = {0.f, 0.f, 0.f, 0.f, 0.f, 0.f, 0.f, 0.f};
    #pragma unroll
    for (int tm = 0; tm < 4; ++tm)
        #pragma unroll
        for (int tn = 0; tn < 4; ++tn) acc[tm][tn] = vzero;

    // Async-stage one k-slab: A 128x32 f16 (512 x 16B chunks, 2/thread) and
    // B 256x32 f16 (1024 x 16B chunks, 4/thread), memory -> LDS, no VGPR bounce.
    auto stage = [&](int kt, int buf) {
        const int kbase = kt * KT;
        #pragma unroll
        for (int s = 0; s < 2; ++s) {
            const int c = tid + s * 256;
            const int r = c >> 2, col = (c & 3) * 8;
            const _Float16* g = nrm + (size_t)(i0 + r) * kD + kbase + col;
            const unsigned  l = (unsigned)(unsigned long long)&As[buf][r][col];
            asm volatile("global_load_async_to_lds_b128 %0, %1, off"
                         :: "v"(l), "v"(g) : "memory");
        }
        #pragma unroll
        for (int s = 0; s < 4; ++s) {
            const int c = tid + s * 256;
            const int r = c >> 2, col = (c & 3) * 8;
            const _Float16* g = nrm + (size_t)(j0 + r) * kD + kbase + col;
            const unsigned  l = (unsigned)(unsigned long long)&Bs[buf][r][col];
            asm volatile("global_load_async_to_lds_b128 %0, %1, off"
                         :: "v"(l), "v"(g) : "memory");
        }
    };

    // A fragment (16x32 f16): lane L holds row M=L%16; halves K = off..off+7
    // and 16+off..16+off+7 with off = (L>=16)*8   -> two ds_load_b128.
    auto loadA = [&](int buf, int tm) -> v16h {
        const int m   = wm * 64 + tm * 16 + (lane & 15);
        const int off = (lane >> 4) * 8;
        const v8h a0 = *(const v8h*)(&As[buf][m][off]);
        const v8h a1 = *(const v8h*)(&As[buf][m][16 + off]);
        v16h a;
        #pragma unroll
        for (int q = 0; q < 8; ++q) { a[q] = a0[q]; a[8 + q] = a1[q]; }
        return a;
    };

    // B fragment (32x16 f16): lane L holds col N=L%16; halves K = kb..kb+7
    // and 16+kb..16+kb+7 with kb = (L>=16)*8       -> two ds_load_b128.
    auto loadB = [&](int buf, int tn) -> v16h {
        const int n  = wn * 64 + tn * 16 + (lane & 15);
        const int kb = (lane >> 4) * 8;
        const v8h b0 = *(const v8h*)(&Bs[buf][n][kb]);
        const v8h b1 = *(const v8h*)(&Bs[buf][n][kb + 16]);
        v16h b;
        #pragma unroll
        for (int q = 0; q < 8; ++q) { b[q] = b0[q]; b[8 + q] = b1[q]; }
        return b;
    };

    stage(0, 0);
    asm volatile("s_wait_asynccnt 0x0" ::: "memory");
    __syncthreads();

    for (int kt = 0; kt < NKT; ++kt) {
        const int cur = kt & 1;
        if (kt + 1 < NKT) stage(kt + 1, cur ^ 1);   // async prefetch next slab

        // Issue all 16 fragment loads first, then a straight 16-WMMA burst:
        // lets the scheduler interleave partial s_wait_dscnt with the WMMAs.
        v16h bfr[4];
        v16h afr[4];
        #pragma unroll
        for (int tn = 0; tn < 4; ++tn) bfr[tn] = loadB(cur, tn);
        #pragma unroll
        for (int tm = 0; tm < 4; ++tm) afr[tm] = loadA(cur, tm);

        #pragma unroll
        for (int tm = 0; tm < 4; ++tm) {
            #pragma unroll
            for (int tn = 0; tn < 4; ++tn) {
                acc[tm][tn] = __builtin_amdgcn_wmma_f32_16x16x32_f16(
                    false, afr[tm], false, bfr[tn], (short)0, acc[tm][tn], false, false);
            }
        }

        asm volatile("s_wait_asynccnt 0x0" ::: "memory");  // next slab landed in LDS
        __syncthreads();                                   // publish across waves
    }

    // ---- fused epilogue: hinge + indicator + triangle mask + reduction ----
    if (tid < BTM) yl[tid] = y[i0 + tid];
    yl[BTM + tid] = y[j0 + tid];          // 256 col labels
    __syncthreads();

    // dist2 = 2 - 2g  (rows normalized)  =>  hinge = max(2g - (2 - theta), 0)
    const float hshift = 2.0f - kTheta;
    float lsum = 0.0f;
    const int mlo = (lane >> 4) * 8;      // C/D layout: M = e + 8*(lane>=16)
    const int nlo = lane & 15;            //             N = lane % 16
    #pragma unroll
    for (int tm = 0; tm < 4; ++tm) {
        #pragma unroll
        for (int tn = 0; tn < 4; ++tn) {
            #pragma unroll
            for (int e = 0; e < 8; ++e) {
                const int li = wm * 64 + tm * 16 + mlo + e;   // local row [0,128)
                const int lj = wn * 64 + tn * 16 + nlo;       // local col [0,256)
                if (j0 + lj >= i0 + li) {
                    const float g     = acc[tm][tn][e];
                    const float hinge = fmaxf(2.0f * g - hshift, 0.0f);
                    const float ind   = (yl[li] == yl[BTM + lj]) ? 1.0f : -1.0f;
                    lsum += kXi + ind * hinge;
                }
            }
        }
    }

    red[tid] = lsum;
    __syncthreads();
    #pragma unroll
    for (int off = 128; off > 0; off >>= 1) {
        if (tid < off) red[tid] += red[tid + off];
        __syncthreads();
    }
    if (tid == 0) {
        const float m = 1.0f / ((float)kB * (float)kB - (float)kB);
        atomicAdd(out, m * red[0]);
    }
}

// ---------------------------------------------------------------------------
extern "C" void kernel_launch(void* const* d_in, const int* in_sizes, int n_in,
                              void* d_out, int out_size, void* d_ws, size_t ws_size,
                              hipStream_t stream) {
    const float* x   = (const float*)d_in[0];
    const int*   yv  = (const int*)d_in[1];
    float*       out = (float*)d_out;
    _Float16*    nrm = (_Float16*)d_ws;      // 8192*512*2 = 8 MB scratch

    zero_out_kernel<<<1, 1, 0, stream>>>(out);
    row_normalize_f16<<<kB, 256, 0, stream>>>(x, nrm);

    dim3 grid(kB / BTN, kB / BTM);           // 32 x 64 blocks; sub-diagonal exits early
    gram_hinge_wmma<<<grid, 256, 0, stream>>>(nrm, yv, out);
}
